// CausalSelfAttention_31164282700033
// MI455X (gfx1250) — compile-verified, matrix operand order checked by reading, not run on hardware
//
#include <hip/hip_runtime.h>

typedef __attribute__((ext_vector_type(16))) _Float16 v16h;
typedef __attribute__((ext_vector_type(8)))  _Float16 v8h;
typedef __attribute__((ext_vector_type(8)))  float    v8f;

#define WMMA_F16(a, b, c) \
  __builtin_amdgcn_wmma_f32_16x16x32_f16(false, (a), false, (b), (short)0, (c), false, false)

// Build v16h A/B fragment for 16x16x32 f16 WMMA from two contiguous 8-half chunks.
// ISA layout (16-bit A 16x32): lane half=0 holds K=0..7 (v0..3) and K=16..23 (v4..7);
// half=1 holds K=8..15 and K=24..31.  So chunk0 = row[k0 + 8*half .. +7],
// chunk1 = row[k0 + 16 + 8*half .. +7].
static __device__ __forceinline__ v16h load_frag_row(const _Float16* rowPtr, int k0, int half) {
  v8h lo = *(const v8h*)(rowPtr + k0 + half * 8);
  v8h hi = *(const v8h*)(rowPtr + k0 + 16 + half * 8);
  return __builtin_shufflevector(lo, hi, 0, 1, 2, 3, 4, 5, 6, 7, 8, 9, 10, 11, 12, 13, 14, 15);
}

static __device__ __forceinline__ v8f zero8f() {
  v8f z = {0.f, 0.f, 0.f, 0.f, 0.f, 0.f, 0.f, 0.f};
  return z;
}

// ---------------------------------------------------------------------------
// Elementwise convert f32 -> f16
__global__ void f32_to_f16_kernel(const float* __restrict__ src, _Float16* __restrict__ dst, int n) {
  int i = blockIdx.x * blockDim.x + threadIdx.x;
  if (i < n) dst[i] = (_Float16)src[i];
}

// Transpose + convert: src[r][c] f32 (rows x cols) -> dst[c][r] f16
__global__ void transpose_f32_to_f16_kernel(const float* __restrict__ src, _Float16* __restrict__ dst,
                                            int rows, int cols) {
  int i = blockIdx.x * blockDim.x + threadIdx.x;
  if (i >= rows * cols) return;
  int r = i / cols, c = i - r * cols;
  dst[(size_t)c * rows + r] = (_Float16)src[i];
}

// ---------------------------------------------------------------------------
// QKV GEMM: qkv = x @ Wqkv + bqkv,  A = xh [4096 x 1024] f16 (row major),
// B given transposed: wT [3072 x 1024] f16.  Each wave computes a 32(M) x 64(N)
// tile with 8 v_wmma_f32_16x16x32_f16 per K-step of 32.
// Epilogue scatters into q[b,h,t,d], k[b,h,t,d], vt[b,h,d,t] (f16).
__global__ void qkv_gemm_kernel(const _Float16* __restrict__ xh,
                                const _Float16* __restrict__ wT,
                                const float* __restrict__ bias,
                                _Float16* __restrict__ qbuf,
                                _Float16* __restrict__ kbuf,
                                _Float16* __restrict__ vtbuf) {
  const int K = 1024;
  const int lane = threadIdx.x & 31;
  const int wave = blockIdx.x * (blockDim.x >> 5) + (threadIdx.x >> 5);
  const int tilesN = 3072 / 64;  // 48
  const int mt = wave / tilesN;  // 0..127 (32-row tiles)
  const int ng = wave % tilesN;
  const int m0 = mt * 32, n0 = ng * 64;
  const int half = lane >> 4, ln = lane & 15;

  const _Float16* arow0 = xh + (size_t)(m0 + ln) * K;
  const _Float16* arow1 = xh + (size_t)(m0 + 16 + ln) * K;
  const _Float16* brow[4];
#pragma unroll
  for (int nt = 0; nt < 4; ++nt) brow[nt] = wT + (size_t)(n0 + nt * 16 + ln) * K;

  v8f acc[2][4];
#pragma unroll
  for (int mi = 0; mi < 2; ++mi)
#pragma unroll
    for (int nt = 0; nt < 4; ++nt) acc[mi][nt] = zero8f();

  for (int k0 = 0; k0 < K; k0 += 32) {
    v16h a0 = load_frag_row(arow0, k0, half);
    v16h a1 = load_frag_row(arow1, k0, half);
#pragma unroll
    for (int nt = 0; nt < 4; ++nt) {
      v16h b = load_frag_row(brow[nt], k0, half);
      acc[0][nt] = WMMA_F16(a0, b, acc[0][nt]);
      acc[1][nt] = WMMA_F16(a1, b, acc[1][nt]);
    }
  }

#pragma unroll
  for (int nt = 0; nt < 4; ++nt) {
    const int c = n0 + nt * 16 + ln;   // 0..3071, fixed per lane for this tile
    const float bv = bias[c];
    const int region = c >> 10;        // 0=q 1=k 2=v
    const int cc = c & 1023;
    const int h = cc >> 6, d = cc & 63;
#pragma unroll
    for (int mi = 0; mi < 2; ++mi) {
#pragma unroll
      for (int r = 0; r < 8; ++r) {
        const int g = m0 + mi * 16 + r + half * 8;  // global row 0..4095
        const int b_ = g >> 11, t = g & 2047;
        const _Float16 hv = (_Float16)(acc[mi][nt][r] + bv);
        if (region == 0)
          qbuf[(((size_t)(b_ * 16 + h)) * 2048 + t) * 64 + d] = hv;
        else if (region == 1)
          kbuf[(((size_t)(b_ * 16 + h)) * 2048 + t) * 64 + d] = hv;
        else
          vtbuf[(((size_t)(b_ * 16 + h)) * 64 + d) * 2048 + t] = hv;
      }
    }
  }
}

// ---------------------------------------------------------------------------
// Flash attention, block-cooperative: one block = 8 waves = 128 query rows of a
// single (b,h).  Each 32-key tile of K and Vt is staged global->LDS ONCE per
// block (256 threads x one 16B chunk each for K and for V), then all 8 waves
// build WMMA fragments from LDS (ds_load_b128, padded strides to avoid bank
// conflicts).  Online softmax in f32 with wave32 shfl_xor butterflies.
// LDS K row stride 72 halves, V row stride 40 halves (both 16B multiples).
#define KSTRIDE 72
#define VSTRIDE 40
__global__ void attn_kernel(const _Float16* __restrict__ qbuf,
                            const _Float16* __restrict__ kbuf,
                            const _Float16* __restrict__ vtbuf,
                            const int* __restrict__ mask,
                            _Float16* __restrict__ yh) {
  __shared__ _Float16 lds_k[32 * KSTRIDE];  // 4.5 KB: K tile, [key 0..31][d 0..63]
  __shared__ _Float16 lds_v[64 * VSTRIDE];  // 5.0 KB: Vt tile, [d 0..63][key 0..31]
  __shared__ _Float16 pbuf[8][16 * 32];     // 8.0 KB: per-wave P staging

  const int tid = threadIdx.x;
  const int lane = tid & 31;
  const int wib = tid >> 5;
  const int wave = blockIdx.x * 8 + wib;
  const int bh = wave >> 7;  // 128 row-tiles per (b,h); all 8 waves share bh
  const int b = bh >> 4, h = bh & 15;
  const int t0 = (wave & 127) * 16;
  const int half = lane >> 4, ln = lane & 15;
  const float scale = 0.125f;  // 1/sqrt(64)

  const _Float16* qb = qbuf + (size_t)bh * 2048 * 64;
  const _Float16* kb = kbuf + (size_t)bh * 2048 * 64;
  const _Float16* vb = vtbuf + (size_t)bh * 64 * 2048;
  const int* mrow = mask + b * 2048;

  // staging assignment: each thread owns one 16B chunk of K and one of V
  const int krow = tid >> 3, kch = tid & 7;   // K: 32 rows x 8 chunks (64 halves)
  const int vrow = tid >> 2, vch = tid & 3;   // V: 64 rows x 4 chunks (32 halves)
  const _Float16* gK = kb + (size_t)krow * 64 + kch * 8;     // + j0*64 per tile
  const _Float16* gV = vb + (size_t)vrow * 2048 + vch * 8;   // + j0 per tile
  _Float16* lK = lds_k + krow * KSTRIDE + kch * 8;
  _Float16* lV = lds_v + vrow * VSTRIDE + vch * 8;

  const _Float16* qrow = qb + (size_t)(t0 + ln) * 64;
  const v16h qfa = load_frag_row(qrow, 0, half);   // K-dim 0..31
  const v16h qfb = load_frag_row(qrow, 32, half);  // K-dim 32..63

  float mrun[8], lrun[8];
  v8f o[4];
#pragma unroll
  for (int r = 0; r < 8; ++r) { mrun[r] = -3.0e30f; lrun[r] = 0.f; }
#pragma unroll
  for (int nt = 0; nt < 4; ++nt) o[nt] = zero8f();

  _Float16* pw = &pbuf[wib][0];

  for (int j0 = 0; j0 < 2048; j0 += 32) {
    // ---- stage this tile's K and Vt into LDS (one chunk per thread each) ----
    __syncthreads();  // previous iteration's LDS reads complete
    *(v8h*)lK = *(const v8h*)(gK + (size_t)j0 * 64);
    *(v8h*)lV = *(const v8h*)(gV + j0);
    if (j0 + 32 < 2048) {  // prefetch next tile into caches (global_prefetch_b8)
      __builtin_prefetch(gK + (size_t)(j0 + 32) * 64, 0, 0);
      __builtin_prefetch(gV + j0 + 32, 0, 0);
    }
    __syncthreads();  // tile visible to all waves

    // ---- scores for 32 keys (two 16-col WMMA tiles), K frags from LDS ----
    const _Float16* krow0 = lds_k + ln * KSTRIDE;
    const _Float16* krow1 = lds_k + (16 + ln) * KSTRIDE;
    v8f s0 = zero8f(), s1 = zero8f();
    s0 = WMMA_F16(qfa, load_frag_row(krow0, 0, half), s0);
    s0 = WMMA_F16(qfb, load_frag_row(krow0, 32, half), s0);
    s1 = WMMA_F16(qfa, load_frag_row(krow1, 0, half), s1);
    s1 = WMMA_F16(qfb, load_frag_row(krow1, 32, half), s1);

    const bool m0v = mrow[j0 + ln] != 0;
    const bool m1v = mrow[j0 + 16 + ln] != 0;

    // ---- online softmax update (8 rows per lane; row m = r + 8*half) ----
#pragma unroll
    for (int r = 0; r < 8; ++r) {
      float a0 = m0v ? s0[r] * scale : -3.0e30f;
      float a1 = m1v ? s1[r] * scale : -3.0e30f;
      float mx = fmaxf(a0, a1);
      mx = fmaxf(mx, __shfl_xor(mx, 1, 32));
      mx = fmaxf(mx, __shfl_xor(mx, 2, 32));
      mx = fmaxf(mx, __shfl_xor(mx, 4, 32));
      mx = fmaxf(mx, __shfl_xor(mx, 8, 32));
      const float mnew = fmaxf(mrun[r], mx);
      const float alpha = __expf(mrun[r] - mnew);
      const float p0 = __expf(a0 - mnew);
      const float p1 = __expf(a1 - mnew);
      float rs = p0 + p1;
      rs += __shfl_xor(rs, 1, 32);
      rs += __shfl_xor(rs, 2, 32);
      rs += __shfl_xor(rs, 4, 32);
      rs += __shfl_xor(rs, 8, 32);
      lrun[r] = lrun[r] * alpha + rs;
      mrun[r] = mnew;
#pragma unroll
      for (int nt = 0; nt < 4; ++nt) o[nt][r] *= alpha;
      // stage P row-major [m][k32] so it can be reread as an A-fragment
      pw[(r + 8 * half) * 32 + ln] = (_Float16)p0;
      pw[(r + 8 * half) * 32 + 16 + ln] = (_Float16)p1;
    }

    // ---- O += P(16x32) @ V(32x64); P A-frag + V B-frags from LDS ----
    // (DS ops from the same wave complete in order; no extra barrier needed.)
    const v16h pf = load_frag_row(pw + ln * 32, 0, half);
#pragma unroll
    for (int nt = 0; nt < 4; ++nt) {
      const _Float16* vr = lds_v + (nt * 16 + ln) * VSTRIDE;
      o[nt] = WMMA_F16(pf, load_frag_row(vr, 0, half), o[nt]);
    }
  }

  // ---- normalize and store y (f16) in [B,T,C] with C = h*64 + d ----
#pragma unroll
  for (int nt = 0; nt < 4; ++nt) {
    const int d = h * 64 + nt * 16 + ln;
#pragma unroll
    for (int r = 0; r < 8; ++r) {
      const int t = t0 + r + 8 * half;
      const float inv = lrun[r] > 0.f ? 1.0f / lrun[r] : 0.f;
      yh[((size_t)(b * 2048 + t)) * 1024 + d] = (_Float16)(o[nt][r] * inv);
    }
  }
}

// ---------------------------------------------------------------------------
// Output projection: out = y @ Wproj + bproj (f32 out).  Same tiling as QKV.
__global__ void proj_gemm_kernel(const _Float16* __restrict__ yh,
                                 const _Float16* __restrict__ wT,  // [1024][1024]
                                 const float* __restrict__ bias,
                                 float* __restrict__ out) {
  const int K = 1024;
  const int lane = threadIdx.x & 31;
  const int wave = blockIdx.x * (blockDim.x >> 5) + (threadIdx.x >> 5);
  const int tilesN = 1024 / 64;  // 16
  const int mt = wave / tilesN;
  const int ng = wave % tilesN;
  const int m0 = mt * 32, n0 = ng * 64;
  const int half = lane >> 4, ln = lane & 15;

  const _Float16* arow0 = yh + (size_t)(m0 + ln) * K;
  const _Float16* arow1 = yh + (size_t)(m0 + 16 + ln) * K;
  const _Float16* brow[4];
#pragma unroll
  for (int nt = 0; nt < 4; ++nt) brow[nt] = wT + (size_t)(n0 + nt * 16 + ln) * K;

  v8f acc[2][4];
#pragma unroll
  for (int mi = 0; mi < 2; ++mi)
#pragma unroll
    for (int nt = 0; nt < 4; ++nt) acc[mi][nt] = zero8f();

  for (int k0 = 0; k0 < K; k0 += 32) {
    v16h a0 = load_frag_row(arow0, k0, half);
    v16h a1 = load_frag_row(arow1, k0, half);
#pragma unroll
    for (int nt = 0; nt < 4; ++nt) {
      v16h bfrag = load_frag_row(brow[nt], k0, half);
      acc[0][nt] = WMMA_F16(a0, bfrag, acc[0][nt]);
      acc[1][nt] = WMMA_F16(a1, bfrag, acc[1][nt]);
    }
  }

#pragma unroll
  for (int nt = 0; nt < 4; ++nt) {
    const int c = n0 + nt * 16 + ln;
    const float bv = bias[c];
#pragma unroll
    for (int mi = 0; mi < 2; ++mi) {
#pragma unroll
      for (int r = 0; r < 8; ++r) {
        const int g = m0 + mi * 16 + r + half * 8;
        out[(size_t)g * 1024 + c] = acc[mi][nt][r] + bv;
      }
    }
  }
}

// ---------------------------------------------------------------------------
extern "C" void kernel_launch(void* const* d_in, const int* in_sizes, int n_in,
                              void* d_out, int out_size, void* d_ws, size_t ws_size,
                              hipStream_t stream) {
  (void)in_sizes; (void)n_in; (void)out_size; (void)ws_size;
  const float* x     = (const float*)d_in[0];   // [2,2048,1024]
  const int*   mask  = (const int*)d_in[1];     // [2,2048]
  const float* Wqkv  = (const float*)d_in[2];   // [1024,3072]
  const float* bqkv  = (const float*)d_in[3];   // [3072]
  const float* Wproj = (const float*)d_in[4];   // [1024,1024]
  const float* bproj = (const float*)d_in[5];   // [1024]
  float* out = (float*)d_out;                   // [2,2048,1024] f32

  char* ws = (char*)d_ws;
  _Float16* xh     = (_Float16*)(ws + 0);             //  8 MB: [4096][1024]
  _Float16* wqkvT  = (_Float16*)(ws + 8388608);       //  6 MB: [3072][1024]
  _Float16* wprojT = (_Float16*)(ws + 14680064);      //  2 MB: [1024][1024]
  _Float16* qbuf   = (_Float16*)(ws + 16777216);      //  8 MB: [2][16][2048][64]
  _Float16* kbuf   = (_Float16*)(ws + 25165824);      //  8 MB: [2][16][2048][64]
  _Float16* vtbuf  = (_Float16*)(ws + 33554432);      //  8 MB: [2][16][64][2048]
  _Float16* yh     = (_Float16*)(ws + 41943040);      //  8 MB: [4096][1024]

  // 1. convert x to f16
  f32_to_f16_kernel<<<dim3(16384), dim3(256), 0, stream>>>(x, xh, 4096 * 1024);
  // 2. transpose+convert weights (N x K layout so WMMA B-fragments are contiguous)
  transpose_f32_to_f16_kernel<<<dim3(12288), dim3(256), 0, stream>>>(Wqkv, wqkvT, 1024, 3072);
  transpose_f32_to_f16_kernel<<<dim3(4096), dim3(256), 0, stream>>>(Wproj, wprojT, 1024, 1024);
  // 3. QKV GEMM -> q, k, v^T   (128 M-tiles * 48 N-groups = 6144 waves / 8 per block)
  qkv_gemm_kernel<<<dim3(768), dim3(256), 0, stream>>>(xh, wqkvT, bqkv, qbuf, kbuf, vtbuf);
  // 4. flash attention          (32 bh * 128 row-tiles = 4096 waves / 8 per block)
  attn_kernel<<<dim3(512), dim3(256), 0, stream>>>(qbuf, kbuf, vtbuf, mask, yh);
  // 5. projection GEMM -> out   (128 * 16 = 2048 waves / 8 per block)
  proj_gemm_kernel<<<dim3(256), dim3(256), 0, stream>>>(yh, wprojT, bproj, out);
}
